// DGCMIA_86878598464342
// MI455X (gfx1250) — matrix operand order repaired and checked by reading, not compiled, not run on hardware
//
#include <hip/hip_runtime.h>
#include <cstddef>

#define D_DIM  512
#define NB     128
#define THRESH 0.15f
#define SSCALE 5.0f
#define CAND   8

typedef __attribute__((ext_vector_type(16))) __bf16 v16bf;
typedef __attribute__((ext_vector_type(8)))  float  v8f;

struct U32x8 { uint4 lo, hi; };

template<int M>
__device__ __forceinline__ void topk_insert(float v, unsigned idx,
                                            float (&tv)[M], unsigned (&ti)[M]) {
  if (v <= tv[M - 1]) return;
  tv[M - 1] = v; ti[M - 1] = idx;
#pragma unroll
  for (int j = M - 1; j > 0; --j) {
    if (tv[j] > tv[j - 1]) {
      float fv = tv[j]; tv[j] = tv[j - 1]; tv[j - 1] = fv;
      unsigned fi = ti[j]; ti[j] = ti[j - 1]; ti[j - 1] = fi;
    }
  }
}

// ---------------- Stage 0: L2-normalize the 128 query rows (fp32) -------------
__global__ __launch_bounds__(256) void knorm(const float* __restrict__ x,
                                             float* __restrict__ y) {
  int b = blockIdx.x, t = threadIdx.x;
  float v0 = x[b * D_DIM + t];
  float v1 = x[b * D_DIM + 256 + t];
  float ss = v0 * v0 + v1 * v1;
#pragma unroll
  for (int off = 16; off >= 1; off >>= 1) ss += __shfl_xor(ss, off);
  __shared__ float wsum[8];
  if ((t & 31) == 0) wsum[t >> 5] = ss;
  __syncthreads();
  float tot = 0.f;
#pragma unroll
  for (int i = 0; i < 8; ++i) tot += wsum[i];
  float inv = rsqrtf(tot);
  y[b * D_DIM + t]       = v0 * inv;
  y[b * D_DIM + 256 + t] = v1 * inv;
}

// ---- Stage 0b: pack Qn^T into the WMMA B-operand (32x16 bf16) lane layout ----
// pq[((kb*8+qt)*32+lane)*8 + v] = bf16pair( Qn[q=qt*16+(lane&15)][k], [k+1] )
//   with k = kb*32 + (lane>=16 ? 8 : 0) + (v>=4 ? 16 : 0) + 2*(v&3)
__global__ void kpack(const float* __restrict__ qn, unsigned* __restrict__ pq) {
  int t = blockIdx.x * 256 + threadIdx.x;          // 0..32767
  int v    = t & 7;
  int lane = (t >> 3) & 31;
  int qt   = (t >> 8) & 7;
  int kb   = t >> 11;
  int q = qt * 16 + (lane & 15);
  int k = kb * 32 + ((lane & 16) ? 8 : 0) + ((v & 4) ? 16 : 0) + 2 * (v & 3);
  unsigned short ul = __builtin_bit_cast(unsigned short, (__bf16)qn[q * D_DIM + k]);
  unsigned short uh = __builtin_bit_cast(unsigned short, (__bf16)qn[q * D_DIM + k + 1]);
  pq[t] = (unsigned)ul | ((unsigned)uh << 16);
}

// --------- Stage 1 (hot): bf16 WMMA sim + per-chunk top-4 per query -----------
__global__ __launch_bounds__(256) void kmain(const float* __restrict__ imem,
                                             const unsigned* __restrict__ pq,
                                             int N, int nChunks,
                                             float* __restrict__ sav,
                                             unsigned* __restrict__ sai) {
  __shared__ unsigned pq_lds[32768];   // 128 KB: packed B operand (all queries)
  __shared__ float    mv[8 * NB * 4];  // 16 KB: per-wave top4 values
  __shared__ unsigned mi[8 * NB * 4];  // 16 KB: per-wave top4 indices
  const int tid  = threadIdx.x;
  const int lane = tid & 31;
  const int w    = tid >> 5;

  for (int t = tid; t < 32768 / 4; t += 256)
    ((uint4*)pq_lds)[t] = ((const uint4*)pq)[t];
  __syncthreads();

  for (int c = blockIdx.x; c < nChunks; c += gridDim.x) {
    const int rowBase = c * 128 + w * 16;
    int loadRow = rowBase + (lane & 15);
    if (loadRow >= N) loadRow = N - 1;                 // clamp OOB tail loads
    const float4* rp = (const float4*)(imem + (size_t)loadRow * D_DIM);
    const int fo = (lane & 16) ? 2 : 0;                // K-half per A layout

    v8f acc[8];
#pragma unroll
    for (int qt = 0; qt < 8; ++qt) acc[qt] = (v8f){0,0,0,0,0,0,0,0};
    float ss = 0.f;

    // One K-step: cvt A to bf16, accumulate row-norm, 8 WMMAs against LDS-B.
    auto kstep = [&](int kb, float4 f0, float4 f1, float4 f2, float4 f3) {
      ss += f0.x*f0.x + f0.y*f0.y + f0.z*f0.z + f0.w*f0.w
          + f1.x*f1.x + f1.y*f1.y + f1.z*f1.z + f1.w*f1.w
          + f2.x*f2.x + f2.y*f2.y + f2.z*f2.z + f2.w*f2.w
          + f3.x*f3.x + f3.y*f3.y + f3.z*f3.z + f3.w*f3.w;
      v16bf a;
      a[0]=(__bf16)f0.x;  a[1]=(__bf16)f0.y;  a[2]=(__bf16)f0.z;  a[3]=(__bf16)f0.w;
      a[4]=(__bf16)f1.x;  a[5]=(__bf16)f1.y;  a[6]=(__bf16)f1.z;  a[7]=(__bf16)f1.w;
      a[8]=(__bf16)f2.x;  a[9]=(__bf16)f2.y;  a[10]=(__bf16)f2.z; a[11]=(__bf16)f2.w;
      a[12]=(__bf16)f3.x; a[13]=(__bf16)f3.y; a[14]=(__bf16)f3.z; a[15]=(__bf16)f3.w;
#pragma unroll
      for (int qt = 0; qt < 8; ++qt) {
        const uint4* bp = (const uint4*)&pq_lds[(((kb * 8 + qt) * 32) + lane) * 8];
        U32x8 raw; raw.lo = bp[0]; raw.hi = bp[1];
        v16bf bv = __builtin_bit_cast(v16bf, raw);
        acc[qt] = __builtin_amdgcn_wmma_f32_16x16x32_bf16(
            false, a, false, bv, (short)0, acc[qt], false, false);
      }
    };

    // Software-pipelined HBM stream: issue next K-step's loads before compute.
    int i0 = fo;
    float4 f0 = rp[i0], f1 = rp[i0 + 1], f2 = rp[i0 + 4], f3 = rp[i0 + 5];
    for (int kb = 0; kb < 15; ++kb) {
      const int i1 = i0 + 8;
      float4 g0 = rp[i1], g1 = rp[i1 + 1], g2 = rp[i1 + 4], g3 = rp[i1 + 5];
      __builtin_prefetch(rp + i1 + 24, 0, 1);          // ~3 K-steps ahead
      kstep(kb, f0, f1, f2, f3);
      f0 = g0; f1 = g1; f2 = g2; f3 = g3;
      i0 = i1;
    }
    kstep(15, f0, f1, f2, f3);

    // Memory-row inverse norms: lane L and L^16 jointly hold one row's squares.
    float sstot = ss + __shfl_xor(ss, 16);
    float rinv = rsqrtf(sstot);
    float rs[8];
#pragma unroll
    for (int r = 0; r < 8; ++r) {
      float lo = __shfl(rinv, r);
      float hi = __shfl(rinv, r + 8);
      rs[r] = (lane < 16) ? lo : hi;                   // row for D-tile VGPR r
    }
    const int rowOff = rowBase + ((lane < 16) ? 0 : 8);

#pragma unroll
    for (int qt = 0; qt < 8; ++qt) {
      float    tv[4] = {-1e30f, -1e30f, -1e30f, -1e30f};
      unsigned ti[4] = {0xffffffffu, 0xffffffffu, 0xffffffffu, 0xffffffffu};
#pragma unroll
      for (int r = 0; r < 8; ++r) {
        int rv = rowOff + r;
        float s = acc[qt][r] * rs[r];
        if (rv >= N) s = -1e30f;
        topk_insert<4>(s, (unsigned)rv, tv, ti);
      }
      float ov[4]; unsigned oi[4];
#pragma unroll
      for (int j = 0; j < 4; ++j) { ov[j] = __shfl_xor(tv[j], 16); oi[j] = __shfl_xor(ti[j], 16); }
#pragma unroll
      for (int j = 0; j < 4; ++j) topk_insert<4>(ov[j], oi[j], tv, ti);
      if (lane < 16) {
        int q = qt * 16 + lane;
        int b = (w * NB + q) * 4;
#pragma unroll
        for (int j = 0; j < 4; ++j) { mv[b + j] = tv[j]; mi[b + j] = ti[j]; }
      }
    }
    __syncthreads();
    if (tid < NB) {
      int q = tid;
      float    tv[4] = {-1e30f, -1e30f, -1e30f, -1e30f};
      unsigned ti[4] = {0xffffffffu, 0xffffffffu, 0xffffffffu, 0xffffffffu};
      for (int w2 = 0; w2 < 8; ++w2) {
        int b = (w2 * NB + q) * 4;
#pragma unroll
        for (int j = 0; j < 4; ++j) topk_insert<4>(mv[b + j], mi[b + j], tv, ti);
      }
      size_t ob = ((size_t)c * NB + q) * 4;
#pragma unroll
      for (int j = 0; j < 4; ++j) { sav[ob + j] = tv[j]; sai[ob + j] = ti[j]; }
    }
    __syncthreads();
  }
}

// ---- Stage 2: global top-8 merge, exact fp32 re-score, softmax weights ------
__global__ __launch_bounds__(256) void ktopk(const float* __restrict__ sav,
                                             const unsigned* __restrict__ sai,
                                             int nChunks, int N,
                                             const float* __restrict__ qn,
                                             const float* __restrict__ imem,
                                             int* __restrict__ midx,
                                             float* __restrict__ mw,
                                             int* __restrict__ mval) {
  const int q = blockIdx.x;
  const int t = threadIdx.x;
  const int lane = t & 31, w = t >> 5;
  float tv[CAND]; unsigned ti[CAND];
#pragma unroll
  for (int j = 0; j < CAND; ++j) { tv[j] = -1e30f; ti[j] = 0xffffffffu; }
  for (int c = t; c < nChunks; c += 256) {
    size_t b = ((size_t)c * NB + q) * 4;
#pragma unroll
    for (int j = 0; j < 4; ++j) topk_insert<CAND>(sav[b + j], sai[b + j], tv, ti);
  }
#pragma unroll
  for (int off = 16; off >= 1; off >>= 1) {
    float ov[CAND]; unsigned oi[CAND];
#pragma unroll
    for (int j = 0; j < CAND; ++j) { ov[j] = __shfl_xor(tv[j], off); oi[j] = __shfl_xor(ti[j], off); }
#pragma unroll
    for (int j = 0; j < CAND; ++j) topk_insert<CAND>(ov[j], oi[j], tv, ti);
  }
  __shared__ float    lv[8][CAND];
  __shared__ unsigned li[8][CAND];
  __shared__ unsigned cidx[CAND];
  __shared__ float    cex[CAND];
  if (lane == 0) {
#pragma unroll
    for (int j = 0; j < CAND; ++j) { lv[w][j] = tv[j]; li[w][j] = ti[j]; }
  }
  __syncthreads();
  if (t == 0) {
    float fv[CAND]; unsigned fi[CAND];
#pragma unroll
    for (int j = 0; j < CAND; ++j) { fv[j] = lv[0][j]; fi[j] = li[0][j]; }
    for (int w2 = 1; w2 < 8; ++w2)
      for (int j = 0; j < CAND; ++j) topk_insert<CAND>(lv[w2][j], li[w2][j], fv, fi);
#pragma unroll
    for (int j = 0; j < CAND; ++j) cidx[j] = fi[j];
  }
  __syncthreads();
  if (t < CAND) {                               // exact fp32 re-score
    unsigned id = cidx[t];
    float ex = -1e30f;
    if (id != 0xffffffffu && (int)id < N) {
      const float* row = imem + (size_t)id * D_DIM;
      const float* qr  = qn + (size_t)q * D_DIM;
      float s = 0.f, ssq = 0.f;
      for (int d = 0; d < D_DIM; ++d) { float m = row[d]; s += qr[d] * m; ssq += m * m; }
      ex = s * rsqrtf(ssq);
    }
    cex[t] = ex;
  }
  __syncthreads();
  if (t == 0) {
    float v[CAND]; unsigned ix[CAND];
#pragma unroll
    for (int j = 0; j < CAND; ++j) { v[j] = cex[j]; ix[j] = cidx[j]; }
    for (int a = 0; a < CAND - 1; ++a) {        // sort desc, tie -> smaller idx
      int best = a;
      for (int b = a + 1; b < CAND; ++b)
        if (v[b] > v[best] || (v[b] == v[best] && ix[b] < ix[best])) best = b;
      float fv = v[a]; v[a] = v[best]; v[best] = fv;
      unsigned fi = ix[a]; ix[a] = ix[best]; ix[best] = fi;
    }
    float logit[4]; int ok[3];
    logit[0] = SSCALE;
    for (int k = 0; k < 3; ++k) {               // drop rank 0, threshold cands
      float cv = v[k + 1];
      ok[k] = (cv > THRESH) ? 1 : 0;
      logit[k + 1] = ok[k] ? SSCALE * cv : -1e30f;
    }
    float mx = logit[0];
    for (int k = 1; k < 4; ++k) mx = fmaxf(mx, logit[k]);
    float e[4], sum = 0.f;
    for (int k = 0; k < 4; ++k) { e[k] = expf(logit[k] - mx); sum += e[k]; }
    for (int k = 0; k < 3; ++k) {
      float wgt = 1.f - e[k + 1] / sum;
      mw[q * 3 + k]   = ok[k] ? wgt : 0.f;
      midx[q * 3 + k] = ok[k] ? (int)ix[k + 1] : 0;
      mval[q * 3 + k] = ok[k];
    }
  }
}

// ------------------------ Stage 3: assemble the output ------------------------
__device__ __forceinline__ int pid_of(int j, const int* __restrict__ mval) {
  if (j < NB) return j;
  int slot = j - NB;
  return mval[slot] ? (slot / 3) : -(slot + 1);
}

__global__ void kfill(const float* __restrict__ ifeat, const float* __restrict__ tfeat,
                      const float* __restrict__ imem,  const float* __restrict__ tmem,
                      const int* __restrict__ midx, const float* __restrict__ mw,
                      const int* __restrict__ mval, float* __restrict__ out) {
  int t = blockIdx.x * 256 + threadIdx.x;
  const int total = 1536 * D_DIM;
  if (t >= total) return;
  int r = t >> 9, c = t & (D_DIM - 1);
  float o = 0.f;
  if (r < 512) {                                        // new_img
    if (r < NB) o = ifeat[(r << 9) | c];
    else { int slot = r - NB; o = mval[slot] ? imem[((size_t)midx[slot] << 9) | c] : 0.f; }
  } else if (r < 1024) {                                // new_txt
    int rr = r - 512;
    if (rr < NB) o = tfeat[(rr << 9) | c];
    else { int slot = rr - NB; o = mval[slot] ? tmem[((size_t)midx[slot] << 9) | c] : 0.f; }
  } else {                                              // labels
    int lr = r - 1024;
    if (lr < NB) {
      if (c == lr) o = 1.f;
      else if (c >= NB && (c - NB) / 3 == lr) o = mw[c - NB];
      else o = 0.f;
    } else {
      o = (pid_of(lr, mval) == pid_of(c, mval)) ? 1.f : 0.f;
    }
  }
  out[t] = o;
}

extern "C" void kernel_launch(void* const* d_in, const int* in_sizes, int n_in,
                              void* d_out, int out_size, void* d_ws, size_t ws_size,
                              hipStream_t stream) {
  const float* ifeat = (const float*)d_in[0];
  const float* tfeat = (const float*)d_in[1];
  const float* imem  = (const float*)d_in[2];
  const float* tmem  = (const float*)d_in[3];
  const int N = in_sizes[2] / D_DIM;
  const int nChunks = (N + 127) / 128;

  char* w = (char*)d_ws;
  float*    qn = (float*)w;                         // 256 KB normalized queries
  unsigned* pq = (unsigned*)(w + 262144);           // 128 KB packed B operand
  size_t off = 262144 + 131072;
  float*    sav = (float*)(w + off);   off += (size_t)nChunks * NB * 4 * sizeof(float);
  unsigned* sai = (unsigned*)(w + off); off += (size_t)nChunks * NB * 4 * sizeof(unsigned);
  int*   midx = (int*)(w + off);   off += NB * 3 * sizeof(int);
  float* mwp  = (float*)(w + off); off += NB * 3 * sizeof(float);
  int*   mval = (int*)(w + off);

  knorm<<<NB, 256, 0, stream>>>(ifeat, qn);
  kpack<<<32768 / 256, 256, 0, stream>>>(qn, pq);
  int grid = nChunks < 1024 ? nChunks : 1024;
  kmain<<<grid, 256, 0, stream>>>(imem, pq, N, nChunks, sav, sai);
  ktopk<<<NB, 256, 0, stream>>>(sav, sai, nChunks, N, qn, imem, midx, mwp, mval);
  const int total = 1536 * D_DIM;
  kfill<<<(total + 255) / 256, 256, 0, stream>>>(ifeat, tfeat, imem, tmem,
                                                 midx, mwp, mval, (float*)d_out);
}
